// HybridAttention_42623255445541
// MI455X (gfx1250) — matrix operand order repaired and checked by reading, not compiled, not run on hardware
//
#include <hip/hip_runtime.h>
#include <hip/hip_bf16.h>

// ---------------------------------------------------------------------------
// HybridAttention for MI455X (gfx1250): bf16 WMMA, fp32 accum, wide bf16
// operand loads, ping-pong software pipelining, async global->LDS V staging.
// B=2 H=16 S=2048 D=1024 dk=64.
// ---------------------------------------------------------------------------

typedef __attribute__((ext_vector_type(16))) __bf16 v16bf;
typedef __attribute__((ext_vector_type(8)))  __bf16 v8bf;
typedef __attribute__((ext_vector_type(8)))  float  v8f;

#define CB 2
#define CH 16
#define CS 2048
#define CD 1024
#define CDK 64
#define CBS (CB*CS)          // 4096
#define CN3 (3*CD)           // 3072
#define NE  (CB*CH*CS*CDK)   // 4,194,304 elements per head-major tensor

// ---- WMMA fragment index maps (wave32, 16-bit operands) -------------------
__device__ __forceinline__ int akmap(int lane, int e) {
    return e + ((lane >> 4) << 4);
}
__device__ __forceinline__ int bkmap(int lane, int e) {
    return e + ((lane >> 4) << 4);
}
// C/D 16x16 f32: vgpr r -> row r + 8*(lane>=16), col = lane&15.

__device__ __forceinline__ v8f wmma_bf(v16bf a, v16bf b, v8f c) {
    return __builtin_amdgcn_wmma_f32_16x16x32_bf16(false, a, false, b,
                                                   (short)0, c, false, false);
}

__device__ __forceinline__ float elu1(float x) {
    return x > 0.0f ? x + 1.0f : __expf(x);   // elu(x)+1
}

// Row-major bf16 A-fragment: p points at row_base + k0 + 16*(lane>>4).
__device__ __forceinline__ v16bf load_a16(const __bf16* __restrict__ p) {
    v8bf lo = *(const v8bf*)p;
    v8bf hi = *(const v8bf*)(p + 8);
    v16bf a;
#pragma unroll
    for (int e = 0; e < 8; ++e) { a[e] = lo[e]; a[8 + e] = hi[e]; }
    return a;
}

// Load 4 consecutive packed B fragments (tiles tn0..tn0+3) at k-tile tk.
__device__ __forceinline__ void load_b4(const __bf16* __restrict__ Wp,
                                        int tk, int tn0, int ntn, int lane,
                                        v16bf bb[4]) {
#pragma unroll
    for (int f = 0; f < 4; ++f)
        bb[f] = *(const v16bf*)(Wp + (((size_t)tk * ntn + tn0 + f) * 32 + lane) * 16);
}

// ---------------------------------------------------------------------------
// one-time converts / packs
// ---------------------------------------------------------------------------
__global__ void ha_cvt_bf16(const float* __restrict__ in, __bf16* __restrict__ out,
                            int n) {
    const size_t i = (size_t)blockIdx.x * blockDim.x + threadIdx.x;
    if (i < (size_t)n) out[i] = (__bf16)in[i];
}

// Pack W [KxN] fp32 into per-fragment-contiguous bf16.
__global__ void ha_pack_w(const float* __restrict__ W, __bf16* __restrict__ Wp,
                          int K, int N) {
    const size_t i = (size_t)blockIdx.x * blockDim.x + threadIdx.x;
    if (i >= (size_t)K * N) return;
    const int e = (int)(i & 15);
    size_t t = i >> 4;
    const int lane = (int)(t & 31); t >>= 5;
    const int ntn = N / 16;
    const int tn = (int)(t % ntn);
    const int tk = (int)(t / ntn);
    const int k = tk * 32 + bkmap(lane, e);
    const int n = tn * 16 + (lane & 15);
    Wp[i] = (__bf16)W[(size_t)k * N + n];
}

// ---------------------------------------------------------------------------
// K1: qkv = x @ Wqkv, scatter into head-major Q/K/V (fp32 + bf16).
// Ping-pong pipelined: fragments for step i+1 loaded during step i's WMMAs.
// ---------------------------------------------------------------------------
__global__ void ha_qkv_gemm(const __bf16* __restrict__ xb,
                            const __bf16* __restrict__ Wqp,
                            float* __restrict__ Qf, float* __restrict__ Kf,
                            float* __restrict__ Vf,
                            __bf16* __restrict__ Qb, __bf16* __restrict__ Kb,
                            __bf16* __restrict__ Vb) {
    const int lane = threadIdx.x & 31;
    const int w    = threadIdx.x >> 5;
    const int m0   = blockIdx.y * 16;
    const int n0   = blockIdx.x * 256 + w * 64;
    const int tn0  = n0 >> 4;
    const int m    = lane & 15;
    const int hi   = lane >> 4;
    const __bf16* arow = xb + (size_t)(m0 + m) * CD + 16 * hi;

    v8f acc[4] = {};
    v16bf aA, aB, bA[4], bB[4];
    aA = load_a16(arow);
    load_b4(Wqp, 0, tn0, CN3 / 16, lane, bA);
    for (int k0 = 0; k0 < CD; k0 += 64) {
        if (k0 + 32 < CD) {                       // prefetch k0+32 -> B set
            aB = load_a16(arow + k0 + 32);
            load_b4(Wqp, (k0 >> 5) + 1, tn0, CN3 / 16, lane, bB);
        }
#pragma unroll
        for (int f = 0; f < 4; ++f) acc[f] = wmma_bf(aA, bA[f], acc[f]);
        if (k0 + 64 < CD) {                       // prefetch k0+64 -> A set
            aA = load_a16(arow + k0 + 64);
            load_b4(Wqp, (k0 >> 5) + 2, tn0, CN3 / 16, lane, bA);
        }
#pragma unroll
        for (int f = 0; f < 4; ++f) acc[f] = wmma_bf(aB, bB[f], acc[f]);
    }
#pragma unroll
    for (int f = 0; f < 4; ++f) {
#pragma unroll
        for (int r = 0; r < 8; ++r) {
            const int row = m0 + r + 8 * hi;           // 0..4095
            const int col = n0 + f * 16 + (lane & 15); // 0..3071
            const int b  = row >> 11;
            const int s  = row & (CS - 1);
            const int rg = col >> 10;
            const int cd = col & (CD - 1);
            const int h  = cd >> 6;
            const int d  = cd & 63;
            const size_t idx = (((size_t)(b * CH + h) * CS) + s) * CDK + d;
            const float v = acc[f][r];
            float*  fd = (rg == 0) ? Qf : (rg == 1) ? Kf : Vf;
            __bf16* bd = (rg == 0) ? Qb : (rg == 1) ? Kb : Vb;
            fd[idx] = v;
            bd[idx] = (__bf16)v;
        }
    }
}

// ---------------------------------------------------------------------------
// K2: flash-style causal SDPA. Block 128 (4 waves) per (qtile, h, b).
// V tile async-staged to LDS overlapping the QK^T WMMAs.
// ---------------------------------------------------------------------------
__global__ void ha_flash_local(const __bf16* __restrict__ Qb,
                               const __bf16* __restrict__ Kb,
                               const __bf16* __restrict__ Vb,
                               float* __restrict__ Alocal) {
    __shared__ __bf16 plds[4][16 * 32];   // P staging (per wave)
    __shared__ __bf16 vtile[32 * 64];     // V tile: 32 keys x 64 dk (4 KB)
    const int lane = threadIdx.x & 31;
    const int w    = threadIdx.x >> 5;
    const int qt   = blockIdx.x;
    const int h    = blockIdx.y;
    const int b    = blockIdx.z;
    const size_t base = (size_t)(b * CH + h) * CS * CDK;
    const int q0 = qt * 64 + w * 16;
    const int m  = lane & 15;
    const int hi = lane >> 4;

    const __bf16* qrow = Qb + base + (size_t)(q0 + m) * CDK + 16 * hi;
    v16bf qa[2];
    qa[0] = load_a16(qrow);
    qa[1] = load_a16(qrow + 32);

    v8f o[4] = {};
    float mrow[8], lrow[8];
#pragma unroll
    for (int r = 0; r < 8; ++r) { mrow[r] = -1e30f; lrow[r] = 0.0f; }
    const float scale = 0.125f;              // 1/sqrt(64)
    const int nsteps = (qt * 64 + 64) / 32;  // uniform over the block

    const unsigned vtile_lds = (unsigned)(size_t)&vtile[0];

    for (int kt = 0; kt < nsteps; ++kt) {
        const int kb = kt * 32;
        // ---- kick off async V-tile copy (32x64 bf16 = 4KB, 16B per lane x2)
        {
            const int t = threadIdx.x;
#pragma unroll
            for (int it = 0; it < 2; ++it) {
                const int eoff = (t + it * 128) * 8;  // element offset in tile
                const unsigned laddr = vtile_lds + (unsigned)(eoff * 2);
                const __bf16* gaddr = Vb + base + (size_t)kb * CDK + eoff;
                asm volatile("global_load_async_to_lds_b128 %0, %1, off"
                             :: "v"(laddr), "v"(gaddr) : "memory");
            }
        }
        // ---- scores: preload all 4 K^T fragments, then 4 WMMAs ----------
        v16bf kf[4];
#pragma unroll
        for (int f = 0; f < 2; ++f) {
            const int n = kb + f * 16 + (lane & 15);  // key index for B frag
            const __bf16* kp = Kb + base + (size_t)n * CDK + 16 * hi;
            kf[f * 2 + 0] = *(const v16bf*)(kp);
            kf[f * 2 + 1] = *(const v16bf*)(kp + 32);
        }
        v8f sc[2] = {};
#pragma unroll
        for (int f = 0; f < 2; ++f) {
            sc[f] = wmma_bf(qa[0], kf[f * 2 + 0], sc[f]);
            sc[f] = wmma_bf(qa[1], kf[f * 2 + 1], sc[f]);
        }
        // ---- online softmax on the 16x32 tile ---------------------------
#pragma unroll
        for (int r = 0; r < 8; ++r) {
            const int qrowi = q0 + r + 8 * hi;
            const int j0 = kb + (lane & 15);
            const int j1 = j0 + 16;
            float s0 = (j0 > qrowi) ? -1e30f : sc[0][r] * scale;
            float s1 = (j1 > qrowi) ? -1e30f : sc[1][r] * scale;
            float mx = fmaxf(s0, s1);
            for (int d2 = 8; d2 >= 1; d2 >>= 1)
                mx = fmaxf(mx, __shfl_xor(mx, d2, 16));
            const float mn = fmaxf(mrow[r], mx);
            const float p0 = __expf(s0 - mn);
            const float p1 = __expf(s1 - mn);
            float ps = p0 + p1;
            for (int d2 = 8; d2 >= 1; d2 >>= 1)
                ps += __shfl_xor(ps, d2, 16);
            const float alpha = __expf(mrow[r] - mn);
            lrow[r] = lrow[r] * alpha + ps;
            mrow[r] = mn;
            const int row = r + 8 * hi;
            plds[w][row * 32 + (lane & 15)]      = (__bf16)p0;
            plds[w][row * 32 + 16 + (lane & 15)] = (__bf16)p1;
#pragma unroll
            for (int f = 0; f < 4; ++f) o[f][r] *= alpha;
        }
        asm volatile("s_wait_asynccnt 0x0" ::: "memory");
        __syncthreads();
        // ---- P (16x32) back as A fragment, V frags from LDS, PV WMMA ----
        v16bf pa;
#pragma unroll
        for (int e = 0; e < 16; ++e)
            pa[e] = plds[w][(lane & 15) * 32 + akmap(lane, e)];
#pragma unroll
        for (int f = 0; f < 4; ++f) {
            v16bf vb;
            const int n = f * 16 + (lane & 15);       // dk column
#pragma unroll
            for (int e = 0; e < 16; ++e)
                vb[e] = vtile[bkmap(lane, e) * CDK + n];
            o[f] = wmma_bf(pa, vb, o[f]);
        }
        __syncthreads();   // before next iteration overwrites vtile/plds
    }
    // epilogue: divide by l, store fp32
#pragma unroll
    for (int f = 0; f < 4; ++f)
#pragma unroll
        for (int r = 0; r < 8; ++r) {
            const int qrowi = q0 + r + 8 * hi;
            const float inv = 1.0f / lrow[r];
            Alocal[base + (size_t)qrowi * CDK + f * 16 + (lane & 15)] = o[f][r] * inv;
        }
}

// ---------------------------------------------------------------------------
// K3a/K3b: memory branch. phi(X)@M with fused denominator (dot with z).
// mode==0: writes A_mem fp32.  mode==1: writes U = V - retr/denom as bf16.
// ---------------------------------------------------------------------------
__global__ void ha_mem_branch(const float* __restrict__ Xf,
                              const float* __restrict__ Mmat,
                              const float* __restrict__ z,
                              const float* __restrict__ Vf,
                              float* __restrict__ AmemOut,
                              __bf16* __restrict__ UOut,
                              int mode) {
    const int lane = threadIdx.x & 31;
    const int w    = threadIdx.x >> 5;
    const int st   = blockIdx.x;
    const int h    = blockIdx.y;
    const int b    = blockIdx.z;
    const int bh   = b * CH + h;
    const size_t base = (size_t)bh * CS * CDK;
    const int s0 = st * 64 + w * 16;
    const int m  = lane & 15;
    const float* Mb = Mmat + (size_t)bh * CDK * CDK;
    const float* zb = z + (size_t)bh * CDK;

    v8f acc[4] = {};
    float partial = 0.0f;
    for (int kk = 0; kk < 2; ++kk) {
        v16bf a;
#pragma unroll
        for (int e = 0; e < 16; ++e) {
            const int k = kk * 32 + akmap(lane, e);
            const float p = elu1(Xf[base + (size_t)(s0 + m) * CDK + k]);
            a[e] = (__bf16)p;
            partial += p * zb[k];
        }
#pragma unroll
        for (int f = 0; f < 4; ++f) {
            v16bf bb;
#pragma unroll
            for (int e = 0; e < 16; ++e)
                bb[e] = (__bf16)Mb[(kk * 32 + bkmap(lane, e)) * CDK + f * 16 + (lane & 15)];
            acc[f] = wmma_bf(a, bb, acc[f]);
        }
    }
    const float dsum = partial + __shfl_xor(partial, 16, 32);
    const int hi = lane >> 4;
#pragma unroll
    for (int r = 0; r < 8; ++r) {
        const int row = r + 8 * hi;
        float den = __shfl(dsum, row, 32);
        den = fmaxf(den, 1e-6f);
#pragma unroll
        for (int f = 0; f < 4; ++f) {
            const size_t idx = base + (size_t)(s0 + row) * CDK + f * 16 + (lane & 15);
            const float v = acc[f][r] / den;
            if (mode == 0) AmemOut[idx] = v;
            else           UOut[idx]    = (__bf16)(Vf[idx] - v);
        }
    }
}

// phiK bf16 (A operand for the delta GEMM)
__global__ void ha_phik(const float* __restrict__ Kf, __bf16* __restrict__ phiKb) {
    const size_t i = (size_t)blockIdx.x * blockDim.x + threadIdx.x;
    if (i < (size_t)NE) phiKb[i] = (__bf16)elu1(Kf[i]);
}

// z_new = z + sum_s phi(K)
__global__ void ha_znew(const float* __restrict__ Kf, const float* __restrict__ z,
                        float* __restrict__ zout) {
    const int d = threadIdx.x;
    const int bh = blockIdx.x;
    float s = z[(size_t)bh * CDK + d];
    const float* kp = Kf + (size_t)bh * CS * CDK + d;
    for (int i = 0; i < CS; ++i) s += elu1(kp[(size_t)i * CDK]);
    zout[(size_t)bh * CDK + d] = s;
}

// ---------------------------------------------------------------------------
// K3c: M_new = M + phiK^T @ U  ([64x2048]@[2048x64] per head).
// ---------------------------------------------------------------------------
__global__ void ha_delta_update(const __bf16* __restrict__ phiKb,
                                const __bf16* __restrict__ Ub,
                                const float* __restrict__ Mmat,
                                float* __restrict__ Mout) {
    const int lane = threadIdx.x & 31;
    const int w    = threadIdx.x >> 5;
    const int bh   = blockIdx.x;
    const size_t base = (size_t)bh * CS * CDK;
    for (int t2 = 0; t2 < 2; ++t2) {
        const int t  = w * 2 + t2;
        const int tm = t >> 2, tn = t & 3;
        v8f acc = {};
        for (int k0 = 0; k0 < CS; k0 += 32) {
            v16bf a, bb;
#pragma unroll
            for (int e = 0; e < 16; ++e)   // A = phiK^T : A[m][k] = phiK[k][m]
                a[e] = phiKb[base + (size_t)(k0 + akmap(lane, e)) * CDK + tm * 16 + (lane & 15)];
#pragma unroll
            for (int e = 0; e < 16; ++e)
                bb[e] = Ub[base + (size_t)(k0 + bkmap(lane, e)) * CDK + tn * 16 + (lane & 15)];
            acc = wmma_bf(a, bb, acc);
        }
        const int hi = lane >> 4;
#pragma unroll
        for (int r = 0; r < 8; ++r) {
            const int row = tm * 16 + r + 8 * hi;
            const int col = tn * 16 + (lane & 15);
            const size_t idx = (size_t)bh * CDK * CDK + (size_t)row * CDK + col;
            Mout[idx] = Mmat[idx] + acc[r];
        }
    }
}

// ---------------------------------------------------------------------------
// K4a: gated combine into bf16 [B,S,D] layout for the output GEMM.
// ---------------------------------------------------------------------------
__global__ void ha_combine(const float* __restrict__ Amem,
                           const float* __restrict__ Alocal,
                           const float* __restrict__ beta,
                           __bf16* __restrict__ Acomb) {
    const size_t i = (size_t)blockIdx.x * blockDim.x + threadIdx.x;
    if (i >= (size_t)NE) return;
    const int d = (int)(i & 63);
    size_t t = i >> 6;
    const int s = (int)(t & (CS - 1)); t >>= 11;
    const int h = (int)(t & (CH - 1));
    const int b = (int)(t >> 4);
    const float g = 1.0f / (1.0f + __expf(-beta[h]));
    const float v = g * Amem[i] + (1.0f - g) * Alocal[i];
    Acomb[((size_t)(b * CS + s)) * CD + h * CDK + d] = (__bf16)v;
}

// ---------------------------------------------------------------------------
// K4b: out = Acomb @ Wout (4096x1024x1024) -> d_out fp32. Pipelined.
// ---------------------------------------------------------------------------
__global__ void ha_out_gemm(const __bf16* __restrict__ Acomb,
                            const __bf16* __restrict__ Wop,
                            float* __restrict__ out) {
    const int lane = threadIdx.x & 31;
    const int w    = threadIdx.x >> 5;
    const int m0   = blockIdx.y * 16;
    const int n0   = blockIdx.x * 256 + w * 64;
    const int tn0  = n0 >> 4;
    const int hi   = lane >> 4;
    const __bf16* arow = Acomb + (size_t)(m0 + (lane & 15)) * CD + 16 * hi;

    v8f acc[4] = {};
    v16bf aA, aB, bA[4], bB[4];
    aA = load_a16(arow);
    load_b4(Wop, 0, tn0, CD / 16, lane, bA);
    for (int k0 = 0; k0 < CD; k0 += 64) {
        if (k0 + 32 < CD) {
            aB = load_a16(arow + k0 + 32);
            load_b4(Wop, (k0 >> 5) + 1, tn0, CD / 16, lane, bB);
        }
#pragma unroll
        for (int f = 0; f < 4; ++f) acc[f] = wmma_bf(aA, bA[f], acc[f]);
        if (k0 + 64 < CD) {
            aA = load_a16(arow + k0 + 64);
            load_b4(Wop, (k0 >> 5) + 2, tn0, CD / 16, lane, bA);
        }
#pragma unroll
        for (int f = 0; f < 4; ++f) acc[f] = wmma_bf(aB, bB[f], acc[f]);
    }
#pragma unroll
    for (int f = 0; f < 4; ++f)
#pragma unroll
        for (int r = 0; r < 8; ++r) {
            const int row = m0 + r + 8 * hi;
            const int col = n0 + f * 16 + (lane & 15);
            out[(size_t)row * CD + col] = acc[f][r];
        }
}

// ---------------------------------------------------------------------------
extern "C" void kernel_launch(void* const* d_in, const int* in_sizes, int n_in,
                              void* d_out, int out_size, void* d_ws, size_t ws_size,
                              hipStream_t stream) {
    const float* x    = (const float*)d_in[0];
    const float* Mmat = (const float*)d_in[1];
    const float* z    = (const float*)d_in[2];
    const float* Wqkv = (const float*)d_in[3];
    const float* Wout = (const float*)d_in[4];
    const float* beta = (const float*)d_in[5];

    float* out    = (float*)d_out;               // [B,S,D] = NE floats
    float* Mout   = out + (size_t)NE;            // [B,H,64,64]
    float* zout   = Mout + (size_t)CB * CH * CDK * CDK;  // [B,H,64]

    // workspace layout
    float* Qf     = (float*)d_ws;
    float* Kf     = Qf + NE;
    float* Vf     = Kf + NE;
    float* Alocal = Vf + NE;
    float* Amem   = Alocal + NE;
    __bf16* Qb    = (__bf16*)(Amem + NE);
    __bf16* Kb    = Qb + NE;
    __bf16* Vb    = Kb + NE;
    __bf16* Ub    = Vb + NE;
    __bf16* phiKb = Ub + NE;
    __bf16* Acomb = phiKb + NE;
    __bf16* xb    = Acomb + NE;                  // [B*S, D] bf16
    __bf16* Wqp   = xb + (size_t)CBS * CD;       // packed Wqkv bf16
    __bf16* Wop   = Wqp + (size_t)CD * CN3;      // packed Wout bf16

    // 0) one-time converts / packs
    ha_cvt_bf16<<<dim3((CBS * CD) / 256), 256, 0, stream>>>(x, xb, CBS * CD);
    ha_pack_w<<<dim3((CD * CN3) / 256), 256, 0, stream>>>(Wqkv, Wqp, CD, CN3);
    ha_pack_w<<<dim3((CD * CD) / 256), 256, 0, stream>>>(Wout, Wop, CD, CD);

    // 1) qkv projection, scatter to head-major fp32 + bf16
    ha_qkv_gemm<<<dim3(CN3 / 256, CBS / 16), 128, 0, stream>>>(
        xb, Wqp, Qf, Kf, Vf, Qb, Kb, Vb);

    // 2) causal flash attention (local branch)
    ha_flash_local<<<dim3(CS / 64, CH, CB), 128, 0, stream>>>(Qb, Kb, Vb, Alocal);

    // 3) memory branch + delta rule
    ha_mem_branch<<<dim3(CS / 64, CH, CB), 128, 0, stream>>>(
        Qf, Mmat, z, nullptr, Amem, nullptr, 0);
    ha_mem_branch<<<dim3(CS / 64, CH, CB), 128, 0, stream>>>(
        Kf, Mmat, z, Vf, nullptr, Ub, 1);
    ha_phik<<<dim3(NE / 256), 256, 0, stream>>>(Kf, phiKb);
    ha_delta_update<<<dim3(CB * CH), 256, 0, stream>>>(phiKb, Ub, Mmat, Mout);
    ha_znew<<<dim3(CB * CH), 64, 0, stream>>>(Kf, z, zout);

    // 4) gated combine + output projection
    ha_combine<<<dim3(NE / 256), 256, 0, stream>>>(Amem, Alocal, beta, Acomb);
    ha_out_gemm<<<dim3(CD / 256, CBS / 16), 128, 0, stream>>>(Acomb, Wop, out);
}
